// AuxPath_74371653697708
// MI455X (gfx1250) — matrix operand order repaired and checked by reading, not compiled, not run on hardware
//
#include <hip/hip_runtime.h>

// ---------------------------------------------------------------------------
// Types for CDNA5 WMMA
// ---------------------------------------------------------------------------
typedef __attribute__((ext_vector_type(16))) __bf16 v16bf;
typedef __attribute__((ext_vector_type(8)))  __bf16 v8bf;
typedef __attribute__((ext_vector_type(8)))  float  v8f;

// ---------------------------------------------------------------------------
// Problem constants
// ---------------------------------------------------------------------------
#define BATCH   8
#define HF      64          // feature map H=W
#define HOUT    256         // scribble / output H=W
#define CIN     768         // 256 + 512 concat
#define CH      256         // HID_CH
#define NCLS    4
#define NPIX    (BATCH*HF*HF)       // 32768 GEMM rows
#define KB      24                  // 768/32 ci-blocks per tap
#define NTAP    9
#define KSTEPS  (NTAP*KB)           // 216

// Workspace layout (bytes)
static constexpr size_t OFF_PIN  = 0;                                   // bf16 [8][66][66][768]
static constexpr size_t SZ_PIN   = (size_t)BATCH*66*66*CIN*2;
static constexpr size_t OFF_WPK  = OFF_PIN + SZ_PIN;                    // bf16 packed B fragments
static constexpr size_t SZ_WPK   = (size_t)NTAP*KB*16*32*16*2;          // 9*24*16 tiles * 512 bf16
static constexpr size_t OFF_X    = OFF_WPK + SZ_WPK;                    // f32 [32768][256] conv out
static constexpr size_t SZ_X     = (size_t)NPIX*CH*4;
static constexpr size_t OFF_AUX  = OFF_X + SZ_X;                        // f32 [32768][256]
static constexpr size_t SZ_AUX   = SZ_X;
static constexpr size_t OFF_LG64 = OFF_AUX + SZ_AUX;                    // f32 [32768][4]
static constexpr size_t SZ_LG64  = (size_t)NPIX*NCLS*4;
static constexpr size_t OFF_SMALL= OFF_LG64 + SZ_LG64;                  // small f32 region

// small-region float indices
#define SI_CHSUM 0      // [256]
#define SI_CHSQ  256    // [256]
#define SI_SCALE 512    // [256]
#define SI_SHIFT 768    // [256]
#define SI_MEMN  1024   // [4*256]
#define SI_FLAGS 2048   // [4]
#define SI_SEMB  2052   // [4*256]
#define SI_WEMB  3076   // [4*256]
#define SI_CNT   4100   // [4]
#define SI_WSUM  4104   // [4]
#define SI_END   4108

// ---------------------------------------------------------------------------
// Helpers
// ---------------------------------------------------------------------------
__device__ __forceinline__ __bf16 f2bf(float f) {
    union { float f; unsigned u; } a; a.f = f;
    unsigned r = a.u + 0x7FFFu + ((a.u >> 16) & 1u);   // round-nearest-even
    union { unsigned short s; __bf16 b; } c;
    c.s = (unsigned short)(r >> 16);
    return c.b;
}
__device__ __forceinline__ int imin(int a, int b) { return a < b ? a : b; }

// ---------------------------------------------------------------------------
// Zero the atomic accumulators (must run first; deterministic across replays)
// ---------------------------------------------------------------------------
__global__ void __launch_bounds__(256) zero_kernel(float* s) {
    int t = threadIdx.x;
    for (int i = t; i < 512; i += 256) s[i] = 0.f;
    for (int i = SI_SEMB + t; i < SI_END; i += 256) s[i] = 0.f;
}

// ---------------------------------------------------------------------------
// Pack concat(feat3,feat4) NCHW f32 -> zero-padded NHWC bf16 [8][66][66][768]
// ---------------------------------------------------------------------------
__global__ void __launch_bounds__(256) pack_input_kernel(
    const float* __restrict__ feat3, const float* __restrict__ feat4,
    __bf16* __restrict__ pin)
{
    size_t o = (size_t)blockIdx.x * 256 + threadIdx.x;
    if (o >= (size_t)BATCH*66*66*CIN) return;
    int c  = (int)(o % CIN);
    size_t q = o / CIN;
    int xp = (int)(q % 66); q /= 66;
    int yp = (int)(q % 66);
    int b  = (int)(q / 66);
    float v = 0.f;
    if (xp >= 1 && xp <= HF && yp >= 1 && yp <= HF) {
        int xi = xp - 1, yi = yp - 1;
        if (c < 256) v = feat3[(((size_t)b*256 +  c      )*HF + yi)*HF + xi];
        else         v = feat4[(((size_t)b*512 + (c-256))*HF + yi)*HF + xi];
    }
    pin[o] = f2bf(v);
}

// ---------------------------------------------------------------------------
// Pack conv_w [co][ci][3][3] f32 -> WMMA B-fragments bf16.
// Fragment layout (16-bit B 32x16): lane L element h ->
//   n = L&15,  k = (h&7) + (L>=16 ? 8 : 0) + (h>=8 ? 16 : 0)
// Linear: (((t*24 + cib)*16 + cot)*32 + L)*16 + h
// ---------------------------------------------------------------------------
__global__ void __launch_bounds__(256) pack_weights_kernel(
    const float* __restrict__ conv_w, __bf16* __restrict__ wpk)
{
    size_t o = (size_t)blockIdx.x * 256 + threadIdx.x;
    if (o >= (size_t)NTAP*KB*16*32*16) return;
    int h   = (int)(o & 15);
    size_t r = o >> 4;
    int L   = (int)(r & 31); r >>= 5;
    int cot = (int)(r & 15); r >>= 4;
    int cib = (int)(r % KB);
    int t   = (int)(r / KB);
    int co  = cot * 16 + (L & 15);
    int k   = (h & 7) + ((L >= 16) ? 8 : 0) + ((h >= 8) ? 16 : 0);
    int ci  = cib * 32 + k;
    int ky  = t / 3, kx = t % 3;
    wpk[o] = f2bf(conv_w[(((size_t)co*CIN + ci)*3 + ky)*3 + kx]);
}

// ---------------------------------------------------------------------------
// Implicit-GEMM 3x3 conv via v_wmma_f32_16x16x32_bf16.
// Wave tile: 64 pixels (one image row) x 64 out-channels -> acc[4][4].
// B fragments staged block-wide through LDS with double-buffered
// GLOBAL_LOAD_ASYNC_TO_LDS_B128 (ASYNCcnt): each K-step needs 4KB of
// pre-packed B; 256 threads each issue one async b128, waves then fill
// fragments from LDS (8x less L2 traffic than per-wave global reads).
// Barrier protocol: the barrier at step s proves all waves finished
// compute(s-1) (the last reader of the buffer being overwritten), so a
// single s_wait_asynccnt + s_barrier per step is sufficient.
// Epilogue: +bias, store x, per-channel sum/sumsq via LDS then global atomics.
// ---------------------------------------------------------------------------
__global__ void __launch_bounds__(256) conv_wmma_kernel(
    const __bf16* __restrict__ pin, const __bf16* __restrict__ wpk,
    const float* __restrict__ conv_b, float* __restrict__ xout,
    float* __restrict__ ch_sum, float* __restrict__ ch_sq)
{
    __shared__ float s_sum[64];
    __shared__ float s_sq[64];
    __shared__ __align__(16) __bf16 s_bb[2][2048];   // 2 x 4KB B tile sets

    int tid = threadIdx.x;
    if (tid < 64) { s_sum[tid] = 0.f; s_sq[tid] = 0.f; }

    int wave = tid >> 5;
    int lane = tid & 31;
    int mg   = blockIdx.x * 8 + wave;       // image-row group: 0..511
    int b    = mg >> 6;
    int y    = mg & 63;
    int coB  = blockIdx.y * 64;

    int mrow = lane & 15;                   // pixel within 16-pixel m-tile
    int kofs = (lane >> 4) * 8;             // A-fragment K half offset

    // async B staging: thread tid moves 16B per K-step; source advances
    // linearly by 16KB per step (s = t*24+cib matches pack order).
    unsigned srcOff  = (unsigned)(blockIdx.y * 4096u) + (unsigned)tid * 16u;
    unsigned ldsDst0 = (unsigned)(uintptr_t)(&s_bb[0][tid * 8]);
    unsigned ldsDst1 = (unsigned)(uintptr_t)(&s_bb[1][tid * 8]);

    // kick off step 0
    asm volatile("global_load_async_to_lds_b128 %0, %1, %2 offset:0"
                 :: "v"(ldsDst0), "v"(srcOff), "s"(wpk) : "memory");

    v8f acc[4][4] = {};                     // [m-tile][n-tile]
    int s = 0;
    for (int ky = 0; ky < 3; ++ky) {
        for (int kx = 0; kx < 3; ++kx) {
            const __bf16* pArow =
                pin + (((size_t)(b*66) + (y + ky))*66 + (mrow + kx))*CIN + kofs;
            for (int cib = 0; cib < KB; ++cib, ++s) {
                // current buffer complete + block-visible
                asm volatile("s_wait_asynccnt 0x0" ::: "memory");
                __syncthreads();
                // launch next step into the other buffer
                if (s < KSTEPS-1) {
                    unsigned dst = (s & 1) ? ldsDst0 : ldsDst1;
                    unsigned off = srcOff + (unsigned)(s + 1) * 16384u;
                    asm volatile("global_load_async_to_lds_b128 %0, %1, %2 offset:0"
                                 :: "v"(dst), "v"(off), "s"(wpk) : "memory");
                }

                const __bf16* pa = pArow + cib * 32;
                v16bf af[4];
                #pragma unroll
                for (int mt = 0; mt < 4; ++mt) {
                    const __bf16* pam = pa + (size_t)mt*16*CIN;
                    v8bf alo = *(const v8bf*)(pam);        // K = kofs..kofs+7
                    v8bf ahi = *(const v8bf*)(pam + 16);   // K = kofs+16..kofs+23
                    af[mt] = __builtin_shufflevector(alo, ahi,
                        0,1,2,3,4,5,6,7,8,9,10,11,12,13,14,15);
                }
                const __bf16* pb = &s_bb[s & 1][lane * 16];
                #pragma unroll
                for (int j = 0; j < 4; ++j) {
                    v16bf bf = *(const v16bf*)(pb + j*512);   // ds_load from LDS
                    #pragma unroll
                    for (int mt = 0; mt < 4; ++mt)
                        acc[mt][j] = __builtin_amdgcn_wmma_f32_16x16x32_bf16(
                            false, af[mt], false, bf, (short)0, acc[mt][j],
                            false, false);
                }
            }
        }
    }

    // Epilogue: C layout -> lane L, vgpr v: M = v + 8*(L>>4), N = L&15
    #pragma unroll
    for (int mt = 0; mt < 4; ++mt) {
        int pix0  = mg*64 + mt*16;
        int mbase = pix0 + ((lane >> 4) * 8);
        #pragma unroll
        for (int j = 0; j < 4; ++j) {
            int co = coB + j*16 + (lane & 15);
            float bv = conv_b[co];
            float lsum = 0.f, lsq = 0.f;
            #pragma unroll
            for (int v = 0; v < 8; ++v) {
                float val = acc[mt][j][v] + bv;
                xout[(size_t)(mbase + v)*CH + co] = val;
                lsum += val; lsq += val*val;
            }
            atomicAdd(&s_sum[j*16 + (lane & 15)], lsum);   // ds_add_f32
            atomicAdd(&s_sq [j*16 + (lane & 15)], lsq);
        }
    }
    __syncthreads();
    if (tid < 64) {
        atomicAdd(&ch_sum[coB + tid], s_sum[tid]);
        atomicAdd(&ch_sq [coB + tid], s_sq[tid]);
    }
}

// ---------------------------------------------------------------------------
// BN stats -> per-channel scale/shift
// ---------------------------------------------------------------------------
__global__ void __launch_bounds__(256) bn_finalize_kernel(
    const float* __restrict__ ch_sum, const float* __restrict__ ch_sq,
    const float* __restrict__ gamma, const float* __restrict__ beta,
    float* __restrict__ scale, float* __restrict__ shift)
{
    int c = threadIdx.x;
    const float N = (float)NPIX;
    float mu  = ch_sum[c] / N;
    float var = ch_sq[c] / N - mu*mu;
    float sc  = gamma[c] * rsqrtf(var + 1e-5f);
    scale[c] = sc;
    shift[c] = beta[c] - mu*sc;
}

// ---------------------------------------------------------------------------
// BN + LeakyReLU -> aux, plus 4-class 1x1 head (wave per pixel)
// ---------------------------------------------------------------------------
__global__ void __launch_bounds__(256) bn_logits_kernel(
    const float* __restrict__ x, const float* __restrict__ scale,
    const float* __restrict__ shift, const float* __restrict__ cls_w,
    float* __restrict__ aux, float* __restrict__ lg64)
{
    int wave = threadIdx.x >> 5, lane = threadIdx.x & 31;
    int pix  = blockIdx.x * 8 + wave;           // 0..32767
    int ch   = lane * 8;
    const float* xp = x + (size_t)pix*CH + ch;
    float yv[8];
    #pragma unroll
    for (int i = 0; i < 8; ++i) {
        float v = scale[ch+i]*xp[i] + shift[ch+i];
        yv[i] = (v >= 0.f) ? v : 0.01f*v;
    }
    float* ap = aux + (size_t)pix*CH + ch;
    #pragma unroll
    for (int i = 0; i < 8; ++i) ap[i] = yv[i];

    float p[NCLS];
    #pragma unroll
    for (int k = 0; k < NCLS; ++k) {
        const float* w = cls_w + (size_t)k*CH + ch;
        float s = 0.f;
        #pragma unroll
        for (int i = 0; i < 8; ++i) s += yv[i]*w[i];
        p[k] = s;
    }
    #pragma unroll
    for (int off = 16; off >= 1; off >>= 1)
        #pragma unroll
        for (int k = 0; k < NCLS; ++k) p[k] += __shfl_xor(p[k], off, 32);
    if (lane == 0)
        for (int k = 0; k < NCLS; ++k) lg64[(size_t)pix*NCLS + k] = p[k];
}

// ---------------------------------------------------------------------------
// Bilinear (align_corners) upsample logits 64->256, NCHW output
// ---------------------------------------------------------------------------
__global__ void __launch_bounds__(256) upsample_kernel(
    const float* __restrict__ lg64, float* __restrict__ out)
{
    int o = blockIdx.x * 256 + threadIdx.x;     // < 8*4*256*256
    if (o >= BATCH*NCLS*HOUT*HOUT) return;
    int xo = o & 255, yo = (o >> 8) & 255, k = (o >> 16) & 3, b = o >> 18;
    const float f = 63.0f / 255.0f;
    float fy = yo * f, fx = xo * f;
    int y0 = (int)fy, x0 = (int)fx;
    int y1 = imin(y0+1, 63), x1 = imin(x0+1, 63);
    float wy = fy - (float)y0, wx = fx - (float)x0;
    size_t base = (size_t)b * 4096;
    float g00 = lg64[(base + y0*64 + x0)*NCLS + k];
    float g01 = lg64[(base + y0*64 + x1)*NCLS + k];
    float g10 = lg64[(base + y1*64 + x0)*NCLS + k];
    float g11 = lg64[(base + y1*64 + x1)*NCLS + k];
    float top = g00*(1.f-wx) + g01*wx;
    float bot = g10*(1.f-wx) + g11*wx;
    out[o] = top*(1.f-wy) + bot*wy;
}

// ---------------------------------------------------------------------------
// argmax(scribble, axis=1) -> float
// ---------------------------------------------------------------------------
__global__ void __launch_bounds__(256) argmax_kernel(
    const int* __restrict__ scribble, float* __restrict__ out)
{
    int o = blockIdx.x * 256 + threadIdx.x;     // < 8*65536
    if (o >= BATCH*HOUT*HOUT) return;
    int b = o >> 16, p = o & 65535;
    const int* s = scribble + (size_t)b*NCLS*65536 + p;
    int best = s[0], idx = 0;
    #pragma unroll
    for (int c = 1; c < NCLS; ++c) {
        int v = s[(size_t)c*65536];
        if (v > best) { best = v; idx = c; }
    }
    out[o] = (float)idx;
}

// ---------------------------------------------------------------------------
// Normalize memory bank rows; flag empty rows. 4 waves, one per class.
// ---------------------------------------------------------------------------
__global__ void __launch_bounds__(128) memnorm_kernel(
    const float* __restrict__ mem, float* __restrict__ memn,
    float* __restrict__ flags)
{
    int c = threadIdx.x >> 5, lane = threadIdx.x & 31;
    int ch = lane * 8;
    float v[8]; float ss = 0.f, nz = 0.f;
    #pragma unroll
    for (int i = 0; i < 8; ++i) {
        v[i] = mem[c*CH + ch + i];
        ss += v[i]*v[i];
        nz += (v[i] != 0.f) ? 1.f : 0.f;
    }
    #pragma unroll
    for (int off = 16; off >= 1; off >>= 1) {
        ss += __shfl_xor(ss, off, 32);
        nz += __shfl_xor(nz, off, 32);
    }
    float norm = sqrtf(ss) + 1e-8f;
    #pragma unroll
    for (int i = 0; i < 8; ++i) memn[c*CH + ch + i] = v[i] / norm;
    if (lane == 0) flags[c] = (nz == 0.f) ? 1.f : 0.f;
}

// ---------------------------------------------------------------------------
// Per-pixel memory statistics on sample 0. Wave per pixel, 32 pixels/wave,
// 256 pixels/block. Bilinear-samples aux[0] on the fly (avoids 67MB af).
// Accumulates sum_emb, (1-cos)-weighted emb_n, counts, wsum into LDS,
// then one global atomic pass per block.
// ---------------------------------------------------------------------------
__global__ void __launch_bounds__(256) mem_accum_kernel(
    const float* __restrict__ aux, const int* __restrict__ scribble,
    const float* __restrict__ memn,
    float* __restrict__ g_semb, float* __restrict__ g_wemb,
    float* __restrict__ g_cnt, float* __restrict__ g_wsum)
{
    __shared__ float s_semb[NCLS*CH];
    __shared__ float s_wemb[NCLS*CH];
    __shared__ float s_cnt[NCLS];
    __shared__ float s_wsum[NCLS];
    int tid = threadIdx.x;
    for (int i = tid; i < NCLS*CH; i += 256) { s_semb[i] = 0.f; s_wemb[i] = 0.f; }
    if (tid < NCLS) { s_cnt[tid] = 0.f; s_wsum[tid] = 0.f; }
    __syncthreads();

    int wave = tid >> 5, lane = tid & 31;
    int ch = lane * 8;
    const float f = 63.0f / 255.0f;

    for (int it = 0; it < 32; ++it) {
        int p  = blockIdx.x * 256 + wave * 32 + it;   // 0..65535
        int yy = p >> 8, xx = p & 255;
        float fy = yy * f, fx = xx * f;
        int y0 = (int)fy, x0 = (int)fx;
        int y1 = imin(y0+1, 63), x1 = imin(x0+1, 63);
        float wy = fy - (float)y0, wx = fx - (float)x0;
        const float* p00 = aux + ((size_t)(y0*64 + x0))*CH + ch;
        const float* p01 = aux + ((size_t)(y0*64 + x1))*CH + ch;
        const float* p10 = aux + ((size_t)(y1*64 + x0))*CH + ch;
        const float* p11 = aux + ((size_t)(y1*64 + x1))*CH + ch;
        float e[8]; float ss = 0.f;
        #pragma unroll
        for (int i = 0; i < 8; ++i) {
            float top = p00[i]*(1.f-wx) + p01[i]*wx;
            float bot = p10[i]*(1.f-wx) + p11[i]*wx;
            e[i] = top*(1.f-wy) + bot*wy;
            ss += e[i]*e[i];
        }
        #pragma unroll
        for (int off = 16; off >= 1; off >>= 1) ss += __shfl_xor(ss, off, 32);
        float norm = sqrtf(ss) + 1e-8f;

        float cosv[NCLS];
        #pragma unroll
        for (int k = 0; k < NCLS; ++k) {
            const float* mn = memn + k*CH + ch;
            float d = 0.f;
            #pragma unroll
            for (int i = 0; i < 8; ++i) d += e[i]*mn[i];
            #pragma unroll
            for (int off = 16; off >= 1; off >>= 1) d += __shfl_xor(d, off, 32);
            cosv[k] = d / norm;
        }

        #pragma unroll
        for (int k = 0; k < NCLS; ++k) {
            int mv = scribble[(size_t)k*65536 + p];   // sample 0
            if (mv == 1) {
                float w = 1.0f - cosv[k];
                #pragma unroll
                for (int i = 0; i < 8; ++i) {
                    atomicAdd(&s_semb[k*CH + ch + i], e[i]);
                    atomicAdd(&s_wemb[k*CH + ch + i], w * e[i] / norm);
                }
                if (lane == 0) {
                    atomicAdd(&s_cnt[k], 1.0f);
                    atomicAdd(&s_wsum[k], w);
                }
            }
        }
    }
    __syncthreads();
    for (int i = tid; i < NCLS*CH; i += 256) {
        atomicAdd(&g_semb[i], s_semb[i]);
        atomicAdd(&g_wemb[i], s_wemb[i]);
    }
    if (tid < NCLS) {
        atomicAdd(&g_cnt[tid],  s_cnt[tid]);
        atomicAdd(&g_wsum[tid], s_wsum[tid]);
    }
}

// ---------------------------------------------------------------------------
// Final EMA / mean update, new_bank, and logits_memory = new_bank @ cls_w.T
// ---------------------------------------------------------------------------
__global__ void __launch_bounds__(256) mem_final_kernel(
    const float* __restrict__ mem, const float* __restrict__ memn,
    const float* __restrict__ flags,
    const float* __restrict__ g_semb, const float* __restrict__ g_wemb,
    const float* __restrict__ g_cnt, const float* __restrict__ g_wsum,
    const float* __restrict__ cls_w, const int* __restrict__ stepp,
    float* __restrict__ out_logmem, float* __restrict__ out_bank)
{
    __shared__ float s_nb[NCLS*CH];
    int tid = threadIdx.x;           // channel
    float step = (float)stepp[0];
    float m = powf(1.0f - step/30000.0f, 0.9f) * 0.9f;
    for (int c = 0; c < NCLS; ++c) {
        float cnt = g_cnt[c];
        float mean_upd = g_semb[c*CH + tid] / fmaxf(cnt, 1.0f);
        float cos_upd  = g_wemb[c*CH + tid] / (g_wsum[c] + 1e-8f);
        float ema = (1.0f - m)*memn[c*CH + tid] + m*cos_upd;
        float upd = (flags[c] > 0.5f) ? mean_upd : ema;
        float nb  = (cnt > 0.0f) ? upd : mem[c*CH + tid];
        s_nb[c*CH + tid] = nb;
        out_bank[c*CH + tid] = nb;
    }
    __syncthreads();
    if (tid < NCLS*NCLS) {
        int c = tid >> 2, k = tid & 3;
        float s = 0.f;
        for (int ch = 0; ch < CH; ++ch) s += s_nb[c*CH + ch]*cls_w[k*CH + ch];
        out_logmem[c*NCLS + k] = s;
    }
}

// ---------------------------------------------------------------------------
// Launch
// ---------------------------------------------------------------------------
extern "C" void kernel_launch(void* const* d_in, const int* in_sizes, int n_in,
                              void* d_out, int out_size, void* d_ws, size_t ws_size,
                              hipStream_t stream) {
    const float* feat3    = (const float*)d_in[0];
    const float* feat4    = (const float*)d_in[1];
    const float* conv_w   = (const float*)d_in[2];
    const float* conv_b   = (const float*)d_in[3];
    const float* bn_gamma = (const float*)d_in[4];
    const float* bn_beta  = (const float*)d_in[5];
    const float* cls_w    = (const float*)d_in[6];
    const float* mem      = (const float*)d_in[7];
    const int*   scribble = (const int*)d_in[8];
    const int*   step     = (const int*)d_in[9];

    char* ws = (char*)d_ws;
    __bf16* pin  = (__bf16*)(ws + OFF_PIN);
    __bf16* wpk  = (__bf16*)(ws + OFF_WPK);
    float*  xbuf = (float*)(ws + OFF_X);
    float*  aux  = (float*)(ws + OFF_AUX);
    float*  lg64 = (float*)(ws + OFF_LG64);
    float*  sm   = (float*)(ws + OFF_SMALL);

    float* ch_sum = sm + SI_CHSUM;
    float* ch_sq  = sm + SI_CHSQ;
    float* scale  = sm + SI_SCALE;
    float* shift  = sm + SI_SHIFT;
    float* memn   = sm + SI_MEMN;
    float* flags  = sm + SI_FLAGS;
    float* g_semb = sm + SI_SEMB;
    float* g_wemb = sm + SI_WEMB;
    float* g_cnt  = sm + SI_CNT;
    float* g_wsum = sm + SI_WSUM;

    float* out         = (float*)d_out;
    float* out_logits  = out;                     // [8,4,256,256]
    float* out_targets = out + 2097152;           // [8,256,256]
    float* out_logmem  = out + 2621440;           // [4,4]
    float* out_bank    = out + 2621456;           // [4,256]

    zero_kernel<<<1, 256, 0, stream>>>(sm);
    pack_input_kernel<<<(int)(((size_t)BATCH*66*66*CIN + 255)/256), 256, 0, stream>>>(feat3, feat4, pin);
    pack_weights_kernel<<<(int)(((size_t)NTAP*KB*16*32*16 + 255)/256), 256, 0, stream>>>(conv_w, wpk);

    dim3 cgrid(64, 4);   // 64 blocks * 8 waves = 512 row-groups; 4 channel groups
    conv_wmma_kernel<<<cgrid, 256, 0, stream>>>(pin, wpk, conv_b, xbuf, ch_sum, ch_sq);

    bn_finalize_kernel<<<1, 256, 0, stream>>>(ch_sum, ch_sq, bn_gamma, bn_beta, scale, shift);
    bn_logits_kernel<<<NPIX/8, 256, 0, stream>>>(xbuf, scale, shift, cls_w, aux, lg64);
    upsample_kernel<<<(BATCH*NCLS*HOUT*HOUT)/256, 256, 0, stream>>>(lg64, out_logits);
    argmax_kernel<<<(BATCH*HOUT*HOUT)/256, 256, 0, stream>>>(scribble, out_targets);
    memnorm_kernel<<<1, 128, 0, stream>>>(mem, memn, flags);
    mem_accum_kernel<<<256, 256, 0, stream>>>(aux, scribble, memn, g_semb, g_wemb, g_cnt, g_wsum);
    mem_final_kernel<<<1, 256, 0, stream>>>(mem, memn, flags, g_semb, g_wemb, g_cnt, g_wsum,
                                            cls_w, step, out_logmem, out_bank);
}
